// VulnerabilityGNN_39238821216443
// MI455X (gfx1250) — matrix operand order repaired and checked by reading, not compiled, not run on hardware
//
#include <hip/hip_runtime.h>
#include <hip/hip_bf16.h>
#include <math.h>

#define NNODES 20000
#define NEDGES 320000
#define NGRAPH 64
#define FIN    16
#define HID    256
#define EDG    (NEDGES + NNODES)   // edges + self loops
#define EPSN   1e-5f

typedef __attribute__((ext_vector_type(16))) _Float16 v16h;
typedef __attribute__((ext_vector_type(8)))  float    v8f;

__device__ __forceinline__ float gelu_f(float x) {
  return 0.5f * x * (1.0f + erff(x * 0.7071067811865476f));
}
__device__ __forceinline__ float act_f(float x, int act) {
  if (act == 1) return gelu_f(x);
  if (act == 2) return x > 0.f ? x : (expf(x) - 1.f);   // elu
  return x;
}
// order-preserving float <-> uint mapping for exact atomic max
__device__ __forceinline__ unsigned fenc(float f) {
  unsigned u = __float_as_uint(f);
  return (u & 0x80000000u) ? ~u : (u | 0x80000000u);
}
__device__ __forceinline__ float fdec(unsigned u) {
  return __uint_as_float((u & 0x80000000u) ? (u & 0x7fffffffu) : ~u);
}
#define ENC_NEG_INF 0x007fffffu   // fenc(-inf)

__device__ __forceinline__ unsigned pack_h2(float x, float y) {
  union { unsigned u; _Float16 h[2]; } p;
  p.h[0] = (_Float16)x;
  p.h[1] = (_Float16)y;
  return p.u;
}
__device__ __forceinline__ v16h load_frag(const unsigned* p) {
  union { v16h v; unsigned u[8]; } r;
  #pragma unroll
  for (int i = 0; i < 8; ++i) r.u[i] = p[i];
  return r.v;
}

// ---------------------------------------------------------------------------
// WMMA GEMM: C[M,N] = A[M,K] * B[K,N] (+bias).  f32 in memory, f16 fragments,
// f32 accumulate via v_wmma_f32_16x16x32_f16.
// Tiles are staged into LDS *pre-swizzled into WMMA fragment order* so each
// lane reads its 16 halves as contiguous wide ds loads (no scalar u16 gather).
// One wave per 16x16 C tile; 8 waves/block share the A tile; K unrolled by 64
// (2 WMMAs per barrier pair).  Requires M%16==0, N%128==0, K%64==0 (true for
// every call site here).
//
// Fragment order used here (contiguous k), lane = hf*16+lm:
//   A slot i: K = hf*16 + i;                           row = lm
//   B slot i: K = hf*16 + i;                           col = lm
// ---------------------------------------------------------------------------
__global__ __launch_bounds__(256)
void k_wmma_gemm(const float* __restrict__ A, const float* __restrict__ B,
                 const float* __restrict__ bias, float* __restrict__ C,
                 int M, int N, int K) {
  __shared__ unsigned Af[2][32][10];      // [stage][lane][8 dwords + pad]
  __shared__ unsigned Bf[2][8][32][10];   // [stage][wave][lane][8 dwords + pad]
  const int tid  = threadIdx.x;
  const int wv   = tid >> 5;
  const int lane = tid & 31;
  const int hf   = lane >> 4;
  const int lm   = lane & 15;
  const int m0 = blockIdx.x * 16;
  const int n0 = (blockIdx.y * 8 + wv) * 16;
  v8f acc = {};
  for (int k0 = 0; k0 < K; k0 += 64) {
    #pragma unroll
    for (int s = 0; s < 2; ++s) {
      const int kb = k0 + s * 32;
      // A tile 16x32: whole block, one float2 per thread, coalesced rows.
      {
        const int r = tid >> 4;            // 0..15
        const int c = (tid & 15) * 2;      // 0..30 (even)
        const float2 a2 = *(const float2*)(A + (size_t)(m0 + r) * K + kb + c);
        const int tl = (c >> 4) * 16 + r;              // dest lane
        const int ti = c & 15;                         // dest slot (even)
        Af[s][tl][ti >> 1] = pack_h2(a2.x, a2.y);
      }
      // B tile 32x16 per wave: 8 row-pairs per lane, 16-float coalesced rows.
      #pragma unroll
      for (int e = 0; e < 8; ++e) {
        const int rp = e * 4 + (lane >> 4) * 2;        // even row 0..30
        const int c  = lane & 15;
        const float b0 = B[(size_t)(kb + rp)     * N + n0 + c];
        const float b1 = B[(size_t)(kb + rp + 1) * N + n0 + c];
        const int tl = (rp >> 4) * 16 + c;             // dest lane
        Bf[s][wv][tl][(rp & 15) >> 1] = pack_h2(b0, b1);
      }
    }
    __syncthreads();
    #pragma unroll
    for (int s = 0; s < 2; ++s) {
      v16h af = load_frag(&Af[s][lane][0]);
      v16h bf = load_frag(&Bf[s][wv][lane][0]);
      acc = __builtin_amdgcn_wmma_f32_16x16x32_f16(false, af, false, bf,
                                                   (short)0, acc, false, false);
    }
    __syncthreads();
  }
  // epilogue: bias is per-column -> constant per lane; hoist the load.
  const float bv = bias ? bias[n0 + lm] : 0.f;
  float* cp = C + (size_t)(m0 + 8 * hf) * N + n0 + lm;
  #pragma unroll
  for (int r = 0; r < 8; ++r) cp[(size_t)r * N] = acc[r] + bv;
}

// ---------------------------------------------------------------------------
// Utility / elementwise kernels
// ---------------------------------------------------------------------------
__global__ void k_zero(float* p, int n) {
  int i = blockIdx.x * 256 + threadIdx.x;
  if (i < n) p[i] = 0.f;
}
__global__ void k_initu(unsigned* p, int n, unsigned v) {
  int i = blockIdx.x * 256 + threadIdx.x;
  if (i < n) p[i] = v;
}
__global__ void k_scale_bias(float* x, float s, const float* __restrict__ b, int n, int C) {
  int i = blockIdx.x * 256 + threadIdx.x;
  if (i < n) x[i] = x[i] * s + b[i & (C - 1)];   // C is a power of two here (256)
}
__global__ void k_combine(const float* __restrict__ xl, const float* __restrict__ xg,
                          const float* __restrict__ ident, float* __restrict__ out, int n) {
  int i = blockIdx.x * 256 + threadIdx.x;
  if (i < n) out[i] = 0.5f * (xl[i] + xg[i]) + ident[i];
}

// semantic feature attention: aw = sigmoid(tanh(x@w1+b1)@w2+b2); xs = x*aw
__global__ void k_sem(const float* __restrict__ x,
                      const float* __restrict__ w1, const float* __restrict__ b1,
                      const float* __restrict__ w2, const float* __restrict__ b2,
                      float* __restrict__ xs) {
  int i = blockIdx.x * 256 + threadIdx.x;
  if (i >= NNODES) return;
  float xi[FIN];
  #pragma unroll
  for (int k = 0; k < FIN; ++k) xi[k] = x[i * FIN + k];
  float t1[32];
  for (int c = 0; c < 32; ++c) {
    float s = b1[c];
    #pragma unroll
    for (int k = 0; k < FIN; ++k) s += xi[k] * w1[k * 32 + c];
    t1[c] = tanhf(s);
  }
  for (int c = 0; c < FIN; ++c) {
    float s = b2[c];
    for (int k = 0; k < 32; ++k) s += t1[k] * w2[k * FIN + c];
    float aw = 1.f / (1.f + expf(-s));
    xs[i * FIN + c] = xi[c] * aw;
  }
}

// projection: cur = xs @ proj_w + proj_b  (K=16, plain VALU)
__global__ void k_proj(const float* __restrict__ xs, const float* __restrict__ pw,
                       const float* __restrict__ pb, float* __restrict__ cur) {
  int idx = blockIdx.x * 256 + threadIdx.x;
  if (idx >= NNODES * HID) return;
  int i = idx / HID, j = idx % HID;
  float s = pb[j];
  #pragma unroll
  for (int k = 0; k < FIN; ++k) s += xs[i * FIN + k] * pw[k * HID + j];
  cur[idx] = s;
}

// per-graph node counts
__global__ void k_cnt(const int* __restrict__ batch, float* cnt) {
  int i = blockIdx.x * 256 + threadIdx.x;
  if (i < NNODES) atomicAdd(&cnt[batch[i]], 1.f);
}

// ---------------- GraphNorm ----------------
__global__ void k_gn_sum(const float* __restrict__ x, const int* __restrict__ batch,
                         float* gs, int n, int C) {
  int idx = blockIdx.x * 256 + threadIdx.x;
  if (idx >= n) return;
  int i = idx / C, j = idx % C;
  atomicAdd(&gs[batch[i] * C + j], x[idx]);
}
__global__ void k_gn_div(float* gs, const float* __restrict__ cnt, int n, int C) {
  int idx = blockIdx.x * 256 + threadIdx.x;
  if (idx < n) gs[idx] /= fmaxf(cnt[idx / C], 1.f);
}
__global__ void k_gn_center(float* x, const int* __restrict__ batch,
                            const float* __restrict__ mean, const float* __restrict__ ms,
                            float* gv, int n, int C) {
  int idx = blockIdx.x * 256 + threadIdx.x;
  if (idx >= n) return;
  int i = idx / C, j = idx % C;
  float o = x[idx] - mean[batch[i] * C + j] * ms[j];
  x[idx] = o;
  atomicAdd(&gv[batch[i] * C + j], o * o);
}
__global__ void k_gn_fin(float* x, const int* __restrict__ batch,
                         const float* __restrict__ gv, const float* __restrict__ w,
                         const float* __restrict__ b, int act,
                         const float* __restrict__ res, int n, int C) {
  int idx = blockIdx.x * 256 + threadIdx.x;
  if (idx >= n) return;
  int i = idx / C, j = idx % C;
  float v = w[j] * x[idx] * rsqrtf(gv[batch[i] * C + j] + EPSN) + b[j];
  v = act_f(v, act);
  if (res) v += res[idx];
  x[idx] = v;
}

// ---------------- GCN ----------------
__global__ void k_deg(const int* __restrict__ ei1, float* deg) {
  int e = blockIdx.x * 256 + threadIdx.x;
  if (e >= EDG) return;
  int d = e < NEDGES ? ei1[e] : e - NEDGES;
  atomicAdd(&deg[d], 1.f);
}
__global__ void k_deg_fin(float* deg) {
  int i = blockIdx.x * 256 + threadIdx.x;
  if (i >= NNODES) return;
  float d = deg[i];
  deg[i] = d > 0.f ? rsqrtf(d) : 0.f;
}
__global__ void k_gcn_scatter(const int* __restrict__ ei0, const int* __restrict__ ei1,
                              const float* __restrict__ h, const float* __restrict__ dis,
                              float* __restrict__ dst) {
  int idx = blockIdx.x * 256 + threadIdx.x;
  if (idx >= EDG * (HID / 4)) return;
  int e = idx >> 6;
  int j = (idx & 63) << 2;
  int s = e < NEDGES ? ei0[e] : e - NEDGES;
  int d = e < NEDGES ? ei1[e] : e - NEDGES;
  float w = dis[s] * dis[d];
  const float4 hv = *(const float4*)(h + (size_t)s * HID + j);
  float* dp = dst + (size_t)d * HID + j;
  atomicAdd(dp + 0, hv.x * w);
  atomicAdd(dp + 1, hv.y * w);
  atomicAdd(dp + 2, hv.z * w);
  atomicAdd(dp + 3, hv.w * w);
}

// ---------------- GAT ----------------
__global__ void k_asd(const float* __restrict__ h, const float* __restrict__ attS,
                      const float* __restrict__ attD, float* asrc, float* adst,
                      int H) {
  int idx = blockIdx.x * 256 + threadIdx.x;
  if (idx >= NNODES * H) return;
  int i = idx / H, hh = idx % H;
  const float* hp = h + ((size_t)i * H + hh) * HID;
  const float* As = attS + hh * HID;
  const float* Ad = attD + hh * HID;
  float s1 = 0.f, s2 = 0.f;
  for (int k = 0; k < HID; ++k) { s1 += hp[k] * As[k]; s2 += hp[k] * Ad[k]; }
  asrc[idx] = s1;
  adst[idx] = s2;
}
__global__ void k_gat_e1(const int* __restrict__ ei0, const int* __restrict__ ei1,
                         const float* __restrict__ asrc, const float* __restrict__ adst,
                         float* eb, unsigned* menc, int H) {
  int idx = blockIdx.x * 256 + threadIdx.x;
  if (idx >= EDG * H) return;
  int e = idx / H, hh = idx % H;
  int s = e < NEDGES ? ei0[e] : e - NEDGES;
  int d = e < NEDGES ? ei1[e] : e - NEDGES;
  float v = asrc[s * H + hh] + adst[d * H + hh];
  v = v > 0.f ? v : 0.2f * v;           // leaky_relu(0.2)
  eb[idx] = v;
  atomicMax(&menc[d * H + hh], fenc(v));
}
__global__ void k_gat_e2(const int* __restrict__ ei1, float* eb,
                         const unsigned* __restrict__ menc, float* den, int H) {
  int idx = blockIdx.x * 256 + threadIdx.x;
  if (idx >= EDG * H) return;
  int e = idx / H, hh = idx % H;
  int d = e < NEDGES ? ei1[e] : e - NEDGES;
  float ex = expf(eb[idx] - fdec(menc[d * H + hh]));
  eb[idx] = ex;
  atomicAdd(&den[d * H + hh], ex);
}
__global__ void k_gat_e3(const int* __restrict__ ei1, float* eb,
                         const float* __restrict__ den, int H) {
  int idx = blockIdx.x * 256 + threadIdx.x;
  if (idx >= EDG * H) return;
  int e = idx / H, hh = idx % H;
  int d = e < NEDGES ? ei1[e] : e - NEDGES;
  eb[idx] = eb[idx] / den[d * H + hh];   // alpha
}
__global__ void k_gat_scatter(const int* __restrict__ ei0, const int* __restrict__ ei1,
                              const float* __restrict__ h, const float* __restrict__ alpha,
                              float* __restrict__ dst, int H) {
  int idx = blockIdx.x * 256 + threadIdx.x;
  if (idx >= EDG * (HID / 4)) return;
  int e = idx >> 6;
  int j = (idx & 63) << 2;
  int s = e < NEDGES ? ei0[e] : e - NEDGES;
  int d = e < NEDGES ? ei1[e] : e - NEDGES;
  float a0 = 0.f, a1 = 0.f, a2 = 0.f, a3 = 0.f;
  for (int hh = 0; hh < H; ++hh) {
    float a = alpha[(size_t)e * H + hh];
    const float4 hv = *(const float4*)(h + ((size_t)s * H + hh) * HID + j);
    a0 += hv.x * a; a1 += hv.y * a; a2 += hv.z * a; a3 += hv.w * a;
  }
  float* dp = dst + (size_t)d * HID + j;
  atomicAdd(dp + 0, a0);
  atomicAdd(dp + 1, a1);
  atomicAdd(dp + 2, a2);
  atomicAdd(dp + 3, a3);
}

// ---------------- pooling ----------------
__global__ void k_pool_sum(const float* __restrict__ x, const int* __restrict__ batch,
                           float* gfeat, unsigned* pmax) {
  int idx = blockIdx.x * 256 + threadIdx.x;
  if (idx >= NNODES * HID) return;
  int i = idx / HID, j = idx % HID;
  int g = batch[i];
  float v = x[idx];
  atomicAdd(&gfeat[g * (3 * HID) + j], v);
  atomicMax(&pmax[g * HID + j], fenc(v));
}
__global__ void k_pool_mm(float* gfeat, const unsigned* __restrict__ pmax,
                          const float* __restrict__ cnt) {
  int idx = blockIdx.x * 256 + threadIdx.x;
  if (idx >= NGRAPH * HID) return;
  int g = idx / HID, j = idx % HID;
  gfeat[g * (3 * HID) + j] /= fmaxf(cnt[g], 1.f);
  float m = fdec(pmax[idx]);
  if (!(m > -3.0e38f)) m = 0.f;          // isfinite -> 0 (empty segment)
  gfeat[g * (3 * HID) + HID + j] = m;
}
__global__ void k_score1(const float* __restrict__ t1, const float* __restrict__ pw2,
                         const float* __restrict__ pb2, const int* __restrict__ batch,
                         float* scores, unsigned* senc) {
  int i = blockIdx.x * 256 + threadIdx.x;
  if (i >= NNODES) return;
  float s = pb2[0];
  for (int k = 0; k < HID / 2; ++k) s += tanhf(t1[(size_t)i * (HID / 2) + k]) * pw2[k];
  scores[i] = s;
  atomicMax(&senc[batch[i]], fenc(s));
}
__global__ void k_score2(float* scores, const unsigned* __restrict__ senc,
                         const int* __restrict__ batch, float* sden) {
  int i = blockIdx.x * 256 + threadIdx.x;
  if (i >= NNODES) return;
  float es = expf(scores[i] - fdec(senc[batch[i]]));
  scores[i] = es;
  atomicAdd(&sden[batch[i]], es);
}
__global__ void k_pool_attn(const float* __restrict__ x, const int* __restrict__ batch,
                            const float* __restrict__ scores, const float* __restrict__ sden,
                            float* gfeat) {
  int idx = blockIdx.x * 256 + threadIdx.x;
  if (idx >= NNODES * HID) return;
  int i = idx / HID, j = idx % HID;
  int g = batch[i];
  float w = scores[i] / sden[g];
  atomicAdd(&gfeat[g * (3 * HID) + 2 * HID + j], x[idx] * w);
}

// ---------------- classifier ----------------
__global__ __launch_bounds__(256)
void k_ln_gelu(float* X, const float* __restrict__ w, const float* __restrict__ b, int C) {
  __shared__ float sh[256];
  int row = blockIdx.x, tid = threadIdx.x;
  float v0 = (tid < C) ? X[(size_t)row * C + tid] : 0.f;
  float v1 = (tid + 256 < C) ? X[(size_t)row * C + tid + 256] : 0.f;
  sh[tid] = v0 + v1;
  __syncthreads();
  for (int st = 128; st > 0; st >>= 1) {
    if (tid < st) sh[tid] += sh[tid + st];
    __syncthreads();
  }
  float mu = sh[0] / C;
  __syncthreads();
  float d0 = (tid < C) ? v0 - mu : 0.f;
  float d1 = (tid + 256 < C) ? v1 - mu : 0.f;
  sh[tid] = d0 * d0 + d1 * d1;
  __syncthreads();
  for (int st = 128; st > 0; st >>= 1) {
    if (tid < st) sh[tid] += sh[tid + st];
    __syncthreads();
  }
  float inv = rsqrtf(sh[0] / C + EPSN);
  if (tid < C)       X[(size_t)row * C + tid]       = gelu_f(d0 * inv * w[tid] + b[tid]);
  if (tid + 256 < C) X[(size_t)row * C + tid + 256] = gelu_f(d1 * inv * w[tid + 256] + b[tid + 256]);
}
__global__ void k_out(const float* __restrict__ g3, const float* __restrict__ ow,
                      const float* __restrict__ ob, const float* __restrict__ temp,
                      float* out) {
  int idx = threadIdx.x;
  if (idx >= NGRAPH * 2) return;
  int g = idx >> 1, c = idx & 1;
  float s = ob[c];
  for (int k = 0; k < HID / 2; ++k) s += g3[g * (HID / 2) + k] * ow[k * 2 + c];
  out[g * 2 + c] = s / temp[0];
}

// ---------------------------------------------------------------------------
extern "C" void kernel_launch(void* const* d_in, const int* in_sizes, int n_in,
                              void* d_out, int out_size, void* d_ws, size_t ws_size,
                              hipStream_t stream) {
  (void)in_sizes; (void)n_in; (void)out_size; (void)ws_size;
  auto F = [&](int i) { return (const float*)d_in[i]; };
  // inputs in setup_inputs() insertion order (depth-first through params)
  const float* x     = F(0);
  const int*   ei0   = (const int*)d_in[1];
  const int*   ei1   = ei0 + NEDGES;
  const int*   batch = (const int*)d_in[2];
  const float* sem_w1 = F(3),  *sem_b1 = F(4),  *sem_w2 = F(5),  *sem_b2 = F(6);
  const float* proj_w = F(7),  *proj_b = F(8);
  const float* in_w = F(9), *in_b = F(10), *in_ms = F(11);
  // gcn[l]: w,b,norm.w,norm.b,norm.ms at 12+5l
  const float* gat_l_w = F(27), *gat_l_as = F(28), *gat_l_ad = F(29), *gat_l_b = F(30);
  const float* gat_l_nw = F(31), *gat_l_nb = F(32), *gat_l_nms = F(33);
  const float* gat_g_w = F(34), *gat_g_as = F(35), *gat_g_ad = F(36), *gat_g_b = F(37);
  const float* gat_g_nw = F(38), *gat_g_nb = F(39), *gat_g_nms = F(40);
  const float* pool_w1 = F(41), *pool_b1 = F(42), *pool_w2 = F(43), *pool_b2 = F(44);
  // cls[i]: w,b,ln_w,ln_b at 45+4i
  const float* out_w = F(57), *out_b = F(58), *temp = F(59);
  float* out = (float*)d_out;

  // workspace bump allocator
  float* W = (float*)d_ws;
  size_t off = 0;
  auto alloc = [&](size_t n) { float* p = W + off; off += (n + 15) & ~(size_t)15; return p; };
  float* xs    = alloc((size_t)NNODES * FIN);
  float* cur   = alloc((size_t)NNODES * HID);
  float* tmp   = alloc((size_t)NNODES * HID);
  float* ident = alloc((size_t)NNODES * HID);
  float* xl    = alloc((size_t)NNODES * HID);
  float* xg    = alloc((size_t)NNODES * HID);
  float* hbuf  = alloc((size_t)NNODES * 8 * HID);   // GAT h; reused as pool t1
  float* ebuf  = alloc((size_t)EDG * 8);
  float* dseg  = alloc((size_t)NNODES * 8);
  unsigned* menc = (unsigned*)alloc((size_t)NNODES * 8);
  float* asrc  = alloc((size_t)NNODES * 8);
  float* adst  = alloc((size_t)NNODES * 8);
  float* deg   = alloc(NNODES);                      // becomes dis = rsqrt(deg)
  float* gsum  = alloc((size_t)NGRAPH * HID);
  float* gvar  = alloc((size_t)NGRAPH * HID);
  float* cnt   = alloc(NGRAPH);
  float* gfeat = alloc((size_t)NGRAPH * 3 * HID);
  float* g1    = alloc((size_t)NGRAPH * 2 * HID);
  float* g2    = alloc((size_t)NGRAPH * HID);
  float* g3    = alloc((size_t)NGRAPH * HID / 2);
  float* scores = alloc(NNODES);
  unsigned* senc = (unsigned*)alloc(NGRAPH);
  float* sden  = alloc(NGRAPH);
  unsigned* pmax = (unsigned*)alloc((size_t)NGRAPH * HID);

  auto NB = [](long long n) { return dim3((unsigned)((n + 255) / 256)); };
  const dim3 T(256);
  auto zero = [&](float* p, long long n) { k_zero<<<NB(n), T, 0, stream>>>(p, (int)n); };

  auto graph_norm = [&](float* Xb, const float* nw, const float* nb, const float* nms,
                        int act, const float* res) {
    const long long n = (long long)NNODES * HID;
    zero(gsum, (long long)NGRAPH * HID);
    zero(gvar, (long long)NGRAPH * HID);
    k_gn_sum<<<NB(n), T, 0, stream>>>(Xb, batch, gsum, (int)n, HID);
    k_gn_div<<<NB(NGRAPH * HID), T, 0, stream>>>(gsum, cnt, NGRAPH * HID, HID);
    k_gn_center<<<NB(n), T, 0, stream>>>(Xb, batch, gsum, nms, gvar, (int)n, HID);
    k_gn_div<<<NB(NGRAPH * HID), T, 0, stream>>>(gvar, cnt, NGRAPH * HID, HID);
    k_gn_fin<<<NB(n), T, 0, stream>>>(Xb, batch, gvar, nw, nb, act, res, (int)n, HID);
  };

  // ---- semantic attention + projection + input norm ----
  k_sem<<<NB(NNODES), T, 0, stream>>>(x, sem_w1, sem_b1, sem_w2, sem_b2, xs);
  k_proj<<<NB((long long)NNODES * HID), T, 0, stream>>>(xs, proj_w, proj_b, cur);
  zero(cnt, NGRAPH);
  k_cnt<<<NB(NNODES), T, 0, stream>>>(batch, cnt);
  graph_norm(cur, in_w, in_b, in_ms, /*gelu*/1, nullptr);

  // ---- degrees (with self-loops) -> dis ----
  zero(deg, NNODES);
  k_deg<<<NB(EDG), T, 0, stream>>>(ei1, deg);
  k_deg_fin<<<NB(NNODES), T, 0, stream>>>(deg);

  // ---- 3 GCN layers ----
  for (int l = 0; l < 3; ++l) {
    const float* gw  = F(12 + 5 * l);
    const float* gb  = F(13 + 5 * l);
    const float* nw  = F(14 + 5 * l);
    const float* nb  = F(15 + 5 * l);
    const float* nms = F(16 + 5 * l);
    hipMemcpyAsync(ident, cur, (size_t)NNODES * HID * sizeof(float),
                   hipMemcpyDeviceToDevice, stream);
    k_wmma_gemm<<<dim3(NNODES / 16, HID / 128), T, 0, stream>>>(
        cur, gw, nullptr, tmp, NNODES, HID, HID);
    zero(cur, (long long)NNODES * HID);
    k_gcn_scatter<<<NB((long long)EDG * (HID / 4)), T, 0, stream>>>(ei0, ei1, tmp, deg, cur);
    k_scale_bias<<<NB((long long)NNODES * HID), T, 0, stream>>>(
        cur, 1.f, gb, NNODES * HID, HID);
    graph_norm(cur, nw, nb, nms, /*gelu*/1, ident);
  }

  // ---- dual GAT ----
  hipMemcpyAsync(ident, cur, (size_t)NNODES * HID * sizeof(float),
                 hipMemcpyDeviceToDevice, stream);
  auto gat = [&](int H, const float* Wm, const float* attS, const float* attD,
                 const float* bb, const float* nw, const float* nb,
                 const float* nms, float* outb) {
    k_wmma_gemm<<<dim3(NNODES / 16, (H * HID) / 128), T, 0, stream>>>(
        cur, Wm, nullptr, hbuf, NNODES, H * HID, HID);
    k_asd<<<NB((long long)NNODES * H), T, 0, stream>>>(hbuf, attS, attD, asrc, adst, H);
    k_initu<<<NB((long long)NNODES * H), T, 0, stream>>>(menc, NNODES * H, ENC_NEG_INF);
    zero(dseg, (long long)NNODES * H);
    k_gat_e1<<<NB((long long)EDG * H), T, 0, stream>>>(ei0, ei1, asrc, adst, ebuf, menc, H);
    k_gat_e2<<<NB((long long)EDG * H), T, 0, stream>>>(ei1, ebuf, menc, dseg, H);
    k_gat_e3<<<NB((long long)EDG * H), T, 0, stream>>>(ei1, ebuf, dseg, H);
    zero(outb, (long long)NNODES * HID);
    k_gat_scatter<<<NB((long long)EDG * (HID / 4)), T, 0, stream>>>(ei0, ei1, hbuf, ebuf, outb, H);
    k_scale_bias<<<NB((long long)NNODES * HID), T, 0, stream>>>(
        outb, 1.f / H, bb, NNODES * HID, HID);
    graph_norm(outb, nw, nb, nms, /*elu*/2, nullptr);
  };
  gat(4, gat_l_w, gat_l_as, gat_l_ad, gat_l_b, gat_l_nw, gat_l_nb, gat_l_nms, xl);
  gat(8, gat_g_w, gat_g_as, gat_g_ad, gat_g_b, gat_g_nw, gat_g_nb, gat_g_nms, xg);
  k_combine<<<NB((long long)NNODES * HID), T, 0, stream>>>(xl, xg, ident, cur, NNODES * HID);

  // ---- hierarchical pooling ----
  zero(gfeat, (long long)NGRAPH * 3 * HID);
  k_initu<<<NB((long long)NGRAPH * HID), T, 0, stream>>>(pmax, NGRAPH * HID, ENC_NEG_INF);
  k_pool_sum<<<NB((long long)NNODES * HID), T, 0, stream>>>(cur, batch, gfeat, pmax);
  k_pool_mm<<<NB((long long)NGRAPH * HID), T, 0, stream>>>(gfeat, pmax, cnt);
  // t1 = cur @ pool_w1 + pool_b1  (reuse hbuf)
  k_wmma_gemm<<<dim3(NNODES / 16, (HID / 2) / 128), T, 0, stream>>>(
      cur, pool_w1, pool_b1, hbuf, NNODES, HID / 2, HID);
  k_initu<<<NB(NGRAPH), T, 0, stream>>>(senc, NGRAPH, ENC_NEG_INF);
  zero(sden, NGRAPH);
  k_score1<<<NB(NNODES), T, 0, stream>>>(hbuf, pool_w2, pool_b2, batch, scores, senc);
  k_score2<<<NB(NNODES), T, 0, stream>>>(scores, senc, batch, sden);
  k_pool_attn<<<NB((long long)NNODES * HID), T, 0, stream>>>(cur, batch, scores, sden, gfeat);

  // ---- classifier MLP (WMMA GEMM + LayerNorm+GELU) ----
  k_wmma_gemm<<<dim3(NGRAPH / 16, (2 * HID) / 128), T, 0, stream>>>(
      gfeat, F(45), F(46), g1, NGRAPH, 2 * HID, 3 * HID);
  k_ln_gelu<<<dim3(NGRAPH), T, 0, stream>>>(g1, F(47), F(48), 2 * HID);
  k_wmma_gemm<<<dim3(NGRAPH / 16, HID / 128), T, 0, stream>>>(
      g1, F(49), F(50), g2, NGRAPH, HID, 2 * HID);
  k_ln_gelu<<<dim3(NGRAPH), T, 0, stream>>>(g2, F(51), F(52), HID);
  k_wmma_gemm<<<dim3(NGRAPH / 16, (HID / 2) / 128), T, 0, stream>>>(
      g2, F(53), F(54), g3, NGRAPH, HID / 2, HID);
  k_ln_gelu<<<dim3(NGRAPH), T, 0, stream>>>(g3, F(55), F(56), HID / 2);
  k_out<<<dim3(1), dim3(128), 0, stream>>>(g3, out_w, out_b, temp, out);
}
